// GGSD_Diffusion_7112465842243
// MI455X (gfx1250) — matrix operand order repaired and checked by reading, not compiled, not run on hardware
//
#include <hip/hip_runtime.h>

// ---------------- problem constants ----------------
#define N_NODES 50000
#define MPAD    50048                 // N_NODES rounded up to 64 (GEMM M padding)
#define N_EDGES 800000
#define ETOT    (N_EDGES + N_NODES)   // edges + self loops
#define HEADS   8
#define LRELU   0.2f
#define BN_EPS  1e-5f
#define LN10000 9.210340371976184f

// ---------------- WMMA types ----------------
typedef __attribute__((ext_vector_type(16))) __bf16 v16bf;
typedef __attribute__((ext_vector_type(8)))  float  v8f;

__device__ __forceinline__ float silu_f(float x) { return x / (1.0f + __expf(-x)); }

// round-to-nearest-even fp32 -> bf16, two values packed into one u32
__device__ __forceinline__ unsigned pack_bf16(float lo, float hi) {
    unsigned a = __builtin_bit_cast(unsigned, lo);
    unsigned b = __builtin_bit_cast(unsigned, hi);
    a = (a + 0x7FFFu + ((a >> 16) & 1u)) >> 16;
    b = (b + 0x7FFFu + ((b >> 16) & 1u)) >> 16;
    return (a & 0xFFFFu) | (b << 16);
}

// order-preserving encode/decode for float atomic max via u32 atomicMax
__device__ __forceinline__ unsigned enc_f32(float f) {
    unsigned u = __builtin_bit_cast(unsigned, f);
    return (u & 0x80000000u) ? ~u : (u | 0x80000000u);
}
__device__ __forceinline__ float dec_f32(unsigned e) {
    return (e & 0x80000000u) ? __builtin_bit_cast(float, e ^ 0x80000000u)
                             : __builtin_bit_cast(float, ~e);
}

// ---------------- bf16 pre-pack kernels (halve GEMM HBM reads) --------------
// A: fp32 [rows,K] -> u32 [MPAD, K/2], row m padded with zeros for m >= N_NODES
__global__ void packA_kernel(const float* __restrict__ src, unsigned* __restrict__ dst, int K)
{
    int m = blockIdx.x;            // 0..MPAD-1
    int j = threadIdx.x;           // 0..K/2-1
    int Kp = blockDim.x;
    float lo = 0.0f, hi = 0.0f;
    if (m < N_NODES) {
        const float* p = src + (size_t)m * K + 2 * j;
        lo = p[0]; hi = p[1];
    }
    dst[(size_t)m * Kp + j] = pack_bf16(lo, hi);
}
// W: fp32 [K,Ncols] -> transposed packed u32 [Ncols, K/2]
__global__ void packW_kernel(const float* __restrict__ W, unsigned* __restrict__ dst, int Ncols)
{
    int n = blockIdx.x;            // 0..Ncols-1
    int j = threadIdx.x;           // 0..K/2-1
    int Kp = blockDim.x;
    const float* p = W + (size_t)(2 * j) * Ncols + n;
    dst[(size_t)n * Kp + j] = pack_bf16(p[0], p[Ncols]);
}

// ---------------------------------------------------------------------------
// GEMM: C[M,Ncols] = act( A[M,K] @ W[K,Ncols] + bias + addend )
// Inputs pre-packed bf16 pairs: Ap [MPAD,K/2] u32, Wp [Ncols,K/2] u32.
// 128 threads = 4 waves; block tile 64x64; wave tile 32x32 (2x2 WMMA 16x16).
// Tiles staged global->LDS with CDNA5 async-to-LDS copies (ASYNCcnt),
// fragments read from LDS, math via v_wmma_f32_16x16x32_bf16 (f32 accum).
// ---------------------------------------------------------------------------
__global__ __launch_bounds__(128)
void gemm_bf16_wmma(const unsigned* __restrict__ Ap, const unsigned* __restrict__ Wp,
                    const float* __restrict__ bias, const float* __restrict__ addend,
                    float* __restrict__ C, int M, int K, int Ncols, int act)
{
    __shared__ unsigned lds_a[64][17];  // [m][k-pair] packed bf16x2 (stride 17: no bank conflicts)
    __shared__ unsigned lds_b[64][17];  // [n][k-pair] packed bf16x2

    const int Kp   = K >> 1;
    const int tid  = threadIdx.x;
    const int wave = tid >> 5;
    const int lane = tid & 31;
    const int half = lane >> 4;
    const int l16  = lane & 15;
    const int wm   = (wave >> 1) * 32;
    const int wn   = (wave & 1)  * 32;

    const int rowBase = blockIdx.x * 64;
    const int colBase = blockIdx.y * 64;

    const unsigned abase = (unsigned)(unsigned long long)(&lds_a[0][0]);
    const unsigned bbase = (unsigned)(unsigned long long)(&lds_b[0][0]);

    v8f acc[2][2];
    for (int i = 0; i < 2; ++i)
        for (int j = 0; j < 2; ++j)
            for (int r = 0; r < 8; ++r) acc[i][j][r] = 0.0f;

    for (int k0p = 0; k0p < Kp; k0p += 16) {
        // stage A and B tiles (64 x 16 u32 each) straight into LDS, no VGPR round trip
        for (int t = tid; t < 64 * 16; t += 128) {
            int m = t >> 4, j = t & 15;
            unsigned dsta = abase + (unsigned)(m * 17 + j) * 4u;
            unsigned long long ga =
                (unsigned long long)(Ap + (size_t)(rowBase + m) * Kp + k0p + j);
            asm volatile("global_load_async_to_lds_b32 %0, %1, off"
                         :: "v"(dsta), "v"(ga) : "memory");
            unsigned dstb = bbase + (unsigned)(m * 17 + j) * 4u;
            unsigned long long gb =
                (unsigned long long)(Wp + (size_t)(colBase + m) * Kp + k0p + j);
            asm volatile("global_load_async_to_lds_b32 %0, %1, off"
                         :: "v"(dstb), "v"(gb) : "memory");
        }
        asm volatile("s_wait_asynccnt 0x0" ::: "memory");
        __syncthreads();

        union { v16bf v; unsigned u[8]; } afrag[2], bfrag[2];
        for (int i = 0; i < 2; ++i) {
            int m = wm + i * 16 + l16;
            int n = wn + i * 16 + l16;
            for (int v = 0; v < 8; ++v) {
                int ja = ((v & 4) ? 8 : 0) + half * 4 + (v & 3);  // A K-pair layout
                afrag[i].u[v] = lds_a[m][ja];
                int jb = ((v & 4) ? 8 : 0) + half * 4 + (v & 3);  // B K-pair layout
                bfrag[i].u[v] = lds_b[n][jb];
            }
        }
        for (int i = 0; i < 2; ++i)
            for (int j = 0; j < 2; ++j)
                acc[i][j] = __builtin_amdgcn_wmma_f32_16x16x32_bf16(
                    false, afrag[i].v, false, bfrag[j].v,
                    (short)0, acc[i][j], false, false);
        __syncthreads();
    }

    // epilogue: bias + optional residual + optional SiLU
    for (int i = 0; i < 2; ++i) {
        for (int j = 0; j < 2; ++j) {
            int col = colBase + wn + j * 16 + l16;
            float bv = bias ? bias[col] : 0.0f;
            for (int r = 0; r < 8; ++r) {
                int row = rowBase + wm + i * 16 + half * 8 + r;
                if (row < M) {
                    float v = acc[i][j][r] + bv;
                    if (addend) v += addend[(size_t)row * Ncols + col];
                    if (act == 1) v = silu_f(v);
                    C[(size_t)row * Ncols + col] = v;
                }
            }
        }
    }
}

// ---------------- sinusoidal position embedding (cos/sin halves swapped) ----
__global__ void posembed_kernel(const float* __restrict__ t, float* __restrict__ emb)
{
    int idx = blockIdx.x * blockDim.x + threadIdx.x;   // n*64 + i
    if (idx >= N_NODES * 64) return;
    int n = idx >> 6, i = idx & 63;
    float freq = __expf(-LN10000 * (float)i / 64.0f);
    float x = t[n] * freq;
    emb[(size_t)n * 128 + i]      = __sinf(x);
    emb[(size_t)n * 128 + 64 + i] = __cosf(x);
}

// ---------------- attention alpha precursors: al_s / al_d [N,H] -------------
__global__ void att_logits_kernel(const float* __restrict__ xw,
                                  const float* __restrict__ att_src,
                                  const float* __restrict__ att_dst,
                                  float* __restrict__ al_s, float* __restrict__ al_d,
                                  int dout)
{
    int idx = blockIdx.x * blockDim.x + threadIdx.x;   // n*8 + h
    if (idx >= N_NODES * HEADS) return;
    int n = idx >> 3, h = idx & 7;
    int Cc = dout / HEADS;
    const float* xp = xw + (size_t)n * dout + h * Cc;
    const float* as = att_src + h * Cc;
    const float* ad = att_dst + h * Cc;
    float s = 0.0f, d = 0.0f;
    for (int c = 0; c < Cc; ++c) { float v = xp[c]; s += v * as[c]; d += v * ad[c]; }
    al_s[idx] = s; al_d[idx] = d;
}

__global__ void fill_zero_kernel(float* __restrict__ p, long long n)
{
    long long i = (long long)blockIdx.x * blockDim.x + threadIdx.x;
    long long stride = (long long)gridDim.x * blockDim.x;
    for (; i < n; i += stride) p[i] = 0.0f;
}

// ---------------- edge pass 1: logits + segment max -------------------------
__global__ void edge_max_kernel(const int* __restrict__ ei,
                                const float* __restrict__ al_s, const float* __restrict__ al_d,
                                float* __restrict__ logits, unsigned* __restrict__ mmax)
{
    int idx = blockIdx.x * blockDim.x + threadIdx.x;   // e*8 + h
    if (idx >= ETOT * HEADS) return;
    int e = idx >> 3, h = idx & 7;
    int s, d;
    if (e < N_EDGES) { s = ei[e]; d = ei[N_EDGES + e]; } else { s = d = e - N_EDGES; }
    float l = al_s[s * 8 + h] + al_d[d * 8 + h];
    l = (l > 0.0f) ? l : LRELU * l;
    logits[idx] = l;
    atomicMax(&mmax[d * 8 + h], enc_f32(l));
}

// ---------------- edge pass 2: exp(logit - max) + denominator ---------------
__global__ void edge_exp_kernel(const int* __restrict__ ei,
                                float* __restrict__ logits, const unsigned* __restrict__ mmax,
                                float* __restrict__ denom)
{
    int idx = blockIdx.x * blockDim.x + threadIdx.x;
    if (idx >= ETOT * HEADS) return;
    int e = idx >> 3, h = idx & 7;
    int d = (e < N_EDGES) ? ei[N_EDGES + e] : (e - N_EDGES);
    float m = dec_f32(mmax[d * 8 + h]);
    float ev = __expf(logits[idx] - m);
    logits[idx] = ev;
    atomicAdd(&denom[d * 8 + h], ev);
}

// ---------------- edge pass 3: agg[dst] += alpha * xw[src] ------------------
// grid.x = edge, block.x = dout channels (coalesced per-channel scatter)
__global__ void edge_aggr_kernel(const int* __restrict__ ei,
                                 const float* __restrict__ evals, const float* __restrict__ denom,
                                 const float* __restrict__ xw, float* __restrict__ agg,
                                 int dout)
{
    int e = blockIdx.x;
    int j = threadIdx.x;
    int Cc = dout / HEADS;
    int h = j / Cc;
    int s, d;
    if (e < N_EDGES) { s = ei[e]; d = ei[N_EDGES + e]; } else { s = d = e - N_EDGES; }
    float alpha = evals[e * 8 + h] / denom[d * 8 + h];
    atomicAdd(&agg[(size_t)d * dout + j], alpha * xw[(size_t)s * dout + j]);
}

// ---------------- x = agg + gat_bias + residual -----------------------------
__global__ void combine_kernel(const float* __restrict__ agg, const float* __restrict__ bias,
                               const float* __restrict__ res, float* __restrict__ x, int dout)
{
    size_t idx = (size_t)blockIdx.x * dout + threadIdx.x;
    x[idx] = agg[idx] + bias[threadIdx.x] + res[idx];
}

// ---------------- BN training stats: one block per channel ------------------
__global__ __launch_bounds__(256)
void bn_stats_kernel(const float* __restrict__ x, float* __restrict__ mean,
                     float* __restrict__ var, int dout)
{
    int j = blockIdx.x;
    __shared__ float ssum[256], ssq[256];
    float s = 0.0f, q = 0.0f;
    for (int n = threadIdx.x; n < N_NODES; n += 256) {
        float v = x[(size_t)n * dout + j];
        s += v; q += v * v;
    }
    ssum[threadIdx.x] = s; ssq[threadIdx.x] = q;
    __syncthreads();
    for (int st = 128; st > 0; st >>= 1) {
        if (threadIdx.x < st) {
            ssum[threadIdx.x] += ssum[threadIdx.x + st];
            ssq[threadIdx.x]  += ssq[threadIdx.x + st];
        }
        __syncthreads();
    }
    if (threadIdx.x == 0) {
        float mu = ssum[0] / (float)N_NODES;
        mean[j] = mu;
        var[j]  = ssq[0] / (float)N_NODES - mu * mu;
    }
}

// ---------------- BN normalize + SiLU (in place) ----------------------------
__global__ void bn_silu_kernel(float* __restrict__ x, const float* __restrict__ mean,
                               const float* __restrict__ var, const float* __restrict__ g,
                               const float* __restrict__ b, int dout)
{
    int j = threadIdx.x;
    size_t idx = (size_t)blockIdx.x * dout + j;
    float v = x[idx];
    v = g[j] * (v - mean[j]) * rsqrtf(var[j] + BN_EPS) + b[j];
    x[idx] = silu_f(v);
}

// ===========================================================================
extern "C" void kernel_launch(void* const* d_in, const int* in_sizes, int n_in,
                              void* d_out, int out_size, void* d_ws, size_t ws_size,
                              hipStream_t stream)
{
    (void)in_sizes; (void)n_in; (void)out_size; (void)ws_size;

    const float* subx   = (const float*)d_in[0];
    const int*   ei     = (const int*)  d_in[1];
    const float* noise  = (const float*)d_in[2];
    const float* init_w = (const float*)d_in[3];
    const float* init_b = (const float*)d_in[4];
    const float* te_w1  = (const float*)d_in[5];
    const float* te_b1  = (const float*)d_in[6];
    const float* te_w2  = (const float*)d_in[7];
    const float* te_b2  = (const float*)d_in[8];
    const float* out_w  = (const float*)d_in[33];
    const float* out_b  = (const float*)d_in[34];

    // ---- workspace carve-out ----
    char* ws = (char*)d_ws;
    size_t off = 0;
    auto give = [&](size_t bytes) -> void* {
        void* p = ws + off;
        off += (bytes + 255) & ~(size_t)255;
        return p;
    };
    float*    bufA  = (float*)   give((size_t)N_NODES * 256 * sizeof(float)); // x
    float*    bufB  = (float*)   give((size_t)N_NODES * 256 * sizeof(float)); // xw / emb0
    float*    bufC  = (float*)   give((size_t)N_NODES * 256 * sizeof(float)); // res / emb1
    float*    bufD  = (float*)   give((size_t)N_NODES * 256 * sizeof(float)); // agg / emb2
    unsigned* Apak  = (unsigned*)give((size_t)MPAD * 128 * sizeof(unsigned)); // packed A (K<=256)
    unsigned* Wpak  = (unsigned*)give((size_t)256 * 128 * sizeof(unsigned));  // packed W^T
    float*    ebuf  = (float*)   give((size_t)ETOT * HEADS * sizeof(float));
    float*    alS   = (float*)   give((size_t)N_NODES * HEADS * sizeof(float));
    float*    alD   = (float*)   give((size_t)N_NODES * HEADS * sizeof(float));
    unsigned* mmax  = (unsigned*)give((size_t)N_NODES * HEADS * sizeof(unsigned));
    float*    denom = (float*)   give((size_t)N_NODES * HEADS * sizeof(float));
    float*    mean  = (float*)   give(256 * sizeof(float));
    float*    var   = (float*)   give(256 * sizeof(float));

    auto packA = [&](const float* src, int K) {
        packA_kernel<<<MPAD, K / 2, 0, stream>>>(src, Apak, K);
    };
    auto packW = [&](const float* W, int K, int Ncols) {
        packW_kernel<<<Ncols, K / 2, 0, stream>>>(W, Wpak, Ncols);
    };
    auto gemm = [&](const float* bias, const float* add, float* C,
                    int K, int Ncols, int act) {
        dim3 g((unsigned)((N_NODES + 63) / 64), (unsigned)(Ncols / 64));
        gemm_bf16_wmma<<<g, 128, 0, stream>>>(Apak, Wpak, bias, add, C,
                                              N_NODES, K, Ncols, act);
    };

    // ---- time embedding + init projection ----
    posembed_kernel<<<(N_NODES * 64 + 255) / 256, 256, 0, stream>>>(noise, bufB);
    packA(bufB, 128); packW(te_w1, 128, 128);
    gemm(te_b1, nullptr, bufC, 128, 128, 1);                 // silu
    packA(bufC, 128); packW(te_w2, 128, 128);
    gemm(te_b2, nullptr, bufD, 128, 128, 0);
    packA(subx, 64);  packW(init_w, 64, 128);
    gemm(init_b, bufD, bufA, 64, 128, 0);                    // x = subx@W + b + emb

    // ---- 3 GAT layers ----
    const int dins[3]  = {128, 256, 256};
    const int douts[3] = {256, 256, 128};
    for (int i = 0; i < 3; ++i) {
        int base = 9 + 8 * i;
        const float* gat_w = (const float*)d_in[base + 0];
        const float* att_s = (const float*)d_in[base + 1];
        const float* att_d = (const float*)d_in[base + 2];
        const float* gat_b = (const float*)d_in[base + 3];
        const float* adj_w = (const float*)d_in[base + 4];
        const float* adj_b = (const float*)d_in[base + 5];
        const float* bn_g  = (const float*)d_in[base + 6];
        const float* bn_b  = (const float*)d_in[base + 7];
        int din = dins[i], dout = douts[i];

        packA(bufA, din);                                     // shared by both GEMMs
        packW(adj_w, din, dout);
        gemm(adj_b, nullptr, bufC, din, dout, 0);             // residual
        packW(gat_w, din, dout);
        gemm(nullptr, nullptr, bufB, din, dout, 0);           // xw

        att_logits_kernel<<<(N_NODES * HEADS + 255) / 256, 256, 0, stream>>>(
            bufB, att_s, att_d, alS, alD, dout);

        fill_zero_kernel<<<1024, 256, 0, stream>>>((float*)mmax, (long long)N_NODES * HEADS);
        fill_zero_kernel<<<1024, 256, 0, stream>>>(denom,        (long long)N_NODES * HEADS);
        fill_zero_kernel<<<4096, 256, 0, stream>>>(bufD,         (long long)N_NODES * dout);

        int eThreads = ETOT * HEADS;
        edge_max_kernel<<<(eThreads + 255) / 256, 256, 0, stream>>>(ei, alS, alD, ebuf, mmax);
        edge_exp_kernel<<<(eThreads + 255) / 256, 256, 0, stream>>>(ei, ebuf, mmax, denom);
        edge_aggr_kernel<<<ETOT, dout, 0, stream>>>(ei, ebuf, denom, bufB, bufD, dout);

        combine_kernel<<<N_NODES, dout, 0, stream>>>(bufD, gat_b, bufC, bufA, dout);
        bn_stats_kernel<<<dout, 256, 0, stream>>>(bufA, mean, var, dout);
        bn_silu_kernel<<<N_NODES, dout, 0, stream>>>(bufA, mean, var, bn_g, bn_b, dout);
    }

    // ---- output projection ----
    packA(bufA, 128); packW(out_w, 128, 64);
    gemm(out_b, nullptr, (float*)d_out, 128, 64, 0);
}
